// LiquidLM_16707422781897
// MI455X (gfx1250) — compile-verified, matrix operand order checked by reading, not run on hardware
//
#include <hip/hip_runtime.h>
#include <cstdint>

typedef __attribute__((ext_vector_type(16))) _Float16 v16h;
typedef __attribute__((ext_vector_type(8)))  float    v8f;

// Problem constants (from reference)
#define BB 4
#define TT 1024
#define VV 32000
#define DD 512
#define LL 3
#define DT_C 1.0f
#define EPS_C 1e-5f

#define NWG 32           // scan workgroups: one 16-col tile of D each
#define KCAT (2*DD)      // 1024 = concat(h, s) K dim for scan GEMM
#define KC   (KCAT/32)   // 32 K-chunks of 32 for scan
#define KCH  (DD/32)     // 16 K-chunks for head GEMM

// ---------------------------------------------------------------------------
// K-index helper matching the ISA 16-bit A/B matrix per-lane layout
// (CDNA5 ISA §7.12.2): VGPR v (pair p), lane-half hi ->
//   K = (v>=4?16:0) + (hi?8:0) + (v&3)*2 + p      for one 32-K chunk.
// Equivalently element e (= v*2+p) of the v16h maps to
//   K = off + e          (e < 8,  off = hi*8)
//   K = off + 16 + (e-8) (e >= 8)
// i.e. two contiguous 8-element runs -> lanes can gather A directly.
// ---------------------------------------------------------------------------
__device__ __forceinline__ int kidx(int v, int hi, int p) {
    return (v >= 4 ? 16 : 0) + (hi ? 8 : 0) + ((v & 3) << 1) + p;
}

__device__ __forceinline__ v8f wmma_f16(v16h a, v16h b, v8f c) {
    return __builtin_amdgcn_wmma_f32_16x16x32_f16(
        false, a, false, b, (short)0, c, false, false);
}

// ---------------------------------------------------------------------------
// Kernel 1: embedding gather  x[t][b][d] = embed[ids[b,t]][d]
// ---------------------------------------------------------------------------
__global__ void k_embed(const int* __restrict__ ids,
                        const float* __restrict__ emb,
                        float* __restrict__ x) {
    int idx = blockIdx.x * 256 + threadIdx.x;          // < B*T*D = 2^21
    int d = idx & (DD - 1);
    int b = (idx >> 9) & (BB - 1);
    int t = idx >> 11;
    x[idx] = emb[(size_t)ids[b * TT + t] * DD + d];
}

// ---------------------------------------------------------------------------
// Kernel 2: pack concatenated [W_in ; W_rec] into WMMA-B f16 layout.
// Wpk[(((i*32+nt)*KC)+kc)*512 + lane*16 + e] holds element
//   (n = nt*16 + lane%16, k = kc*32 + kidx(e/2, lane/16, e%2)) of layer i.
// ---------------------------------------------------------------------------
__global__ void k_pack_w(const float* __restrict__ Win,
                         const float* __restrict__ Wrec,
                         _Float16* __restrict__ Wpk) {
    int idx = blockIdx.x * 256 + threadIdx.x;          // < 3*32*32*512
    int e    =  idx        & 15;
    int lane = (idx >>  4) & 31;
    int kc   = (idx >>  9) & 31;
    int nt   = (idx >> 14) & 31;
    int i    =  idx >> 19;
    int n = nt * 16 + (lane & 15);
    int k = kc * 32 + kidx(e >> 1, lane >> 4, e & 1);
    float v = (k < DD) ? Win[((size_t)i * DD + n) * DD + k]
                       : Wrec[((size_t)i * DD + n) * DD + (k - DD)];
    Wpk[idx] = (_Float16)v;
}

// ---------------------------------------------------------------------------
// Kernel 3: pack W_head into WMMA-B f16 layout (does the v-major transpose).
// ---------------------------------------------------------------------------
__global__ void k_pack_whead(const float* __restrict__ Wh,
                             _Float16* __restrict__ WB) {
    int idx = blockIdx.x * 256 + threadIdx.x;          // < 2000*16*512
    int e    =  idx        & 15;
    int lane = (idx >>  4) & 31;
    int kc   = (idx >>  9) & 15;
    int nt   =  idx >> 13;
    int n = nt * 16 + (lane & 15);
    int k = kc * 32 + kidx(e >> 1, lane >> 4, e & 1);
    WB[idx] = (_Float16)Wh[(size_t)n * DD + k];
}

// ---------------------------------------------------------------------------
// cooperative global->LDS copy of B*D floats
// ---------------------------------------------------------------------------
__device__ __forceinline__ void coop_load_hs(float* dst_lds,
                                             const float* src_gbl, int tid) {
    for (int q = tid; q < BB * DD; q += 256) dst_lds[q] = src_gbl[q];
}

// ---------------------------------------------------------------------------
// Device-wide split barrier: monotonic counter, 32 arrivals per phase.
// ---------------------------------------------------------------------------
__device__ __forceinline__ void gbar(unsigned* bar, int tid) {
    __threadfence();                 // flush this thread's global stores
    __syncthreads();
    if (tid == 0) {
        unsigned prev = __hip_atomic_fetch_add(bar, 1u, __ATOMIC_ACQ_REL,
                                               __HIP_MEMORY_SCOPE_AGENT);
        unsigned target = (prev / NWG + 1u) * NWG;
        while (__hip_atomic_load(bar, __ATOMIC_ACQUIRE,
                                 __HIP_MEMORY_SCOPE_AGENT) < target)
            __builtin_amdgcn_s_sleep(1);
    }
    __syncthreads();
}

// ---------------------------------------------------------------------------
// Kernel 4: the liquid-NN recurrent scan.
// 32 WGs x 256 threads. WG nt owns output columns [nt*16, nt*16+16).
// Weights stay resident in VGPRs (12 x v16h per wave) for all 1024 steps.
// Per layer: lanes gather their A operand straight from LDS fp32 staging
// (two contiguous 8-float runs per chunk), WMMA K-split across 8 waves ->
// LDS reduce -> cell update -> exchange h / s slices via L2 -> barrier.
// ---------------------------------------------------------------------------
__global__ __launch_bounds__(256)
void k_scan(const float* __restrict__ x, const _Float16* __restrict__ Wpk,
            const float* __restrict__ bias, const float* __restrict__ tau,
            const float* __restrict__ Avec, const float* __restrict__ gamma,
            const float* __restrict__ beta,
            float* __restrict__ sbufA, float* __restrict__ sbufB,
            float* __restrict__ h_buf, float* __restrict__ hs,
            float* __restrict__ out_state, unsigned* __restrict__ bar) {
    const int tid  = threadIdx.x;
    const int lane = tid & 31;
    const int wv   = tid >> 5;       // 0..7
    const int nt   = blockIdx.x;     // 0..31
    const int m    = lane & 15;      // A-row this lane feeds (batch if < 4)
    const int hi   = lane >> 4;      // lane half -> K sub-offset

    __shared__ float shH[BB * DD];                    // full h (or x_t)   8KB
    __shared__ float shS[BB * DD];                    // full s_cur[i]     8KB
    __shared__ float shC[8 * 256 + 256];              // partials / LN     9KB

    // ---- preload this WG's weight tile into registers (once) ----
    v16h wB[LL][4];
#pragma unroll
    for (int i = 0; i < LL; ++i)
#pragma unroll
        for (int kk = 0; kk < 4; ++kk) {
            const _Float16* p =
                Wpk + ((((size_t)i * 32 + nt) * KC) + (wv * 4 + kk)) * 512 +
                lane * 16;
            wB[i][kk] = *(const v16h*)p;
        }

    for (int t = 0; t < TT; ++t) {
        const float* scur = (t & 1) ? sbufB : sbufA;
        float*       snxt = (t & 1) ? sbufA : sbufB;

#pragma unroll
        for (int i = 0; i < LL; ++i) {
            // gather full h input and full s_cur[i] into LDS
            const float* hsrc = (i == 0) ? (x + (size_t)t * BB * DD) : h_buf;
            coop_load_hs(shH, hsrc, tid);
            coop_load_hs(shS, scur + (size_t)i * BB * DD, tid);
            __syncthreads();

            // K-split matmul: wave wv handles K-chunks [wv*4, wv*4+4).
            // A operand built in registers from shH/shS; rows m>=4 are 0.
            v8f c = {};
#pragma unroll
            for (int kk = 0; kk < 4; ++kk) {
                int kc = wv * 4 + kk;                       // 0..31
                const float* src = (kc < 16) ? shH : shS;   // chunk never
                const float* row =                           // straddles h|s
                    src + m * DD + (kc & 15) * 32 + hi * 8;
                v16h a = {};
                if (m < BB) {
#pragma unroll
                    for (int e = 0; e < 8; ++e) {
                        a[e]     = (_Float16)row[e];
                        a[e + 8] = (_Float16)row[16 + e];
                    }
                }
                c = wmma_f16(a, wB[i][kk], c);
            }
            // stash partial C (16x16 f32), layout: (M = j + hi*8, N = lane%16)
#pragma unroll
            for (int j = 0; j < 8; ++j)
                shC[wv * 256 + (j + hi * 8) * 16 + m] = c[j];
            __syncthreads();

            // reduce 8 partial tiles
            float acc = shC[tid];
#pragma unroll
            for (int w = 1; w < 8; ++w) acc += shC[w * 256 + tid];

            // liquid cell update for this WG's 16 columns, 4 batch rows
            if (tid < BB * 16) {
                int b = tid >> 4, nloc = tid & 15;
                int col = nt * 16 + nloc;
                float pre = acc + bias[i * DD + col];
                float f = tanhf(pre);
                float sold = shS[b * DD + col];
                float a_ = Avec[i * DD + col];
                float tu = tau[i * DD + col];
                float si = (sold + DT_C * a_ * f) /
                           (1.0f + DT_C * (1.0f / tu + f));
                float hn = shH[b * DD + col] + si;
                snxt[((size_t)i * BB + b) * DD + col] = si;
                h_buf[b * DD + col] = hn;
                if (t == TT - 1)
                    out_state[((size_t)i * BB + b) * DD + col] = si;
            }
            gbar(bar, tid);   // all slices of h / s visible device-wide
        }

        // ---- LayerNorm of h, write hs[(b*T + t)*D + col] slice ----
        coop_load_hs(shH, h_buf, tid);
        gbar(bar, tid);       // everyone captured h before next step clobbers
        {
            int b = tid >> 6, u = tid & 63;
            float s1 = 0.f, s2 = 0.f;
            for (int k = u; k < DD; k += 64) {
                float v = shH[b * DD + k];
                s1 += v; s2 += v * v;
            }
            shC[tid] = s1; shC[256 + tid] = s2;
            __syncthreads();
            for (int st = 32; st > 0; st >>= 1) {
                if (u < st) {
                    shC[tid]       += shC[tid + st];
                    shC[256 + tid] += shC[256 + tid + st];
                }
                __syncthreads();
            }
            if (tid < 64) {
                int b2 = tid >> 4, nloc = tid & 15;
                int col = nt * 16 + nloc;
                float mu  = shC[b2 * 64] * (1.0f / DD);
                float var = shC[256 + b2 * 64] * (1.0f / DD) - mu * mu;
                float rs  = rsqrtf(var + EPS_C);
                float v = (shH[b2 * DD + col] - mu) * rs * gamma[col] + beta[col];
                hs[((size_t)b2 * TT + t) * DD + col] = v;
            }
            __syncthreads();
        }
    }
}

// ---------------------------------------------------------------------------
// Kernel 5: pack hs (fp32, rows r = b*T+t) into WMMA-A f16 layout.
// ---------------------------------------------------------------------------
__global__ void k_pack_hs(const float* __restrict__ hs,
                          _Float16* __restrict__ hsA) {
    int idx = blockIdx.x * 256 + threadIdx.x;          // < 256*16*512 = 2^21
    int e    =  idx        & 15;
    int lane = (idx >>  4) & 31;
    int kc   = (idx >>  9) & 15;
    int mt   =  idx >> 13;
    int r = mt * 16 + (lane & 15);
    int k = kc * 32 + kidx(e >> 1, lane >> 4, e & 1);
    hsA[idx] = (_Float16)hs[(size_t)r * DD + k];
}

// ---------------------------------------------------------------------------
// Kernel 6: head GEMM  logits[r, v] = hs[r,:] . W_head[v,:]
// Grid (125, 64) x 256 threads. Wave = 32M x 64N (8 accumulator tiles).
// All operands pre-swizzled -> only aligned b128 loads + v_wmma.
// ---------------------------------------------------------------------------
__global__ __launch_bounds__(256)
void k_gemm_head(const _Float16* __restrict__ hsA,
                 const _Float16* __restrict__ WB,
                 float* __restrict__ out) {
    const int tid  = threadIdx.x;
    const int lane = tid & 31;
    const int wv   = tid >> 5;
    const int wm   = wv >> 2;                 // 0..1
    const int wn   = wv & 3;                  // 0..3
    const int mb0  = blockIdx.y * 4 + wm * 2; // M 16-blocks (2 per wave)
    const int nb0  = blockIdx.x * 16 + wn * 4;// N 16-blocks (4 per wave)

    v8f acc[2][4];
#pragma unroll
    for (int mi = 0; mi < 2; ++mi)
#pragma unroll
        for (int q = 0; q < 4; ++q) { v8f z = {}; acc[mi][q] = z; }

    for (int kc = 0; kc < KCH; ++kc) {
        v16h a0 = *(const v16h*)(hsA + ((size_t)(mb0 + 0) * KCH + kc) * 512 +
                                 lane * 16);
        v16h a1 = *(const v16h*)(hsA + ((size_t)(mb0 + 1) * KCH + kc) * 512 +
                                 lane * 16);
        if (kc + 1 < KCH) {     // L2 prefetch of next A strip
            __builtin_prefetch(hsA + ((size_t)mb0 * KCH + kc + 1) * 512 +
                               lane * 16, 0, 1);
        }
#pragma unroll
        for (int q = 0; q < 4; ++q) {
            v16h b = *(const v16h*)(WB + ((size_t)(nb0 + q) * KCH + kc) * 512 +
                                    lane * 16);
            acc[0][q] = wmma_f16(a0, b, acc[0][q]);
            acc[1][q] = wmma_f16(a1, b, acc[1][q]);
        }
    }

    // epilogue: C layout lane<16 -> (M=j, N=lane), lane>=16 -> (M=j+8)
#pragma unroll
    for (int mi = 0; mi < 2; ++mi)
#pragma unroll
        for (int q = 0; q < 4; ++q) {
            int col = (nb0 + q) * 16 + (lane & 15);
#pragma unroll
            for (int j = 0; j < 8; ++j) {
                int r = (mb0 + mi) * 16 + j + (lane >> 4) * 8;
                out[(size_t)r * VV + col] = acc[mi][q][j];
            }
        }
}

// ---------------------------------------------------------------------------
// Host launcher
// ---------------------------------------------------------------------------
extern "C" void kernel_launch(void* const* d_in, const int* in_sizes, int n_in,
                              void* d_out, int out_size, void* d_ws,
                              size_t ws_size, hipStream_t stream) {
    const int*   ids   = (const int*)  d_in[0];
    const float* emb   = (const float*)d_in[1];
    const float* Win   = (const float*)d_in[2];
    const float* Wrec  = (const float*)d_in[3];
    const float* bias  = (const float*)d_in[4];
    const float* tau   = (const float*)d_in[5];
    const float* Avec  = (const float*)d_in[6];
    const float* gamma = (const float*)d_in[7];
    const float* beta  = (const float*)d_in[8];
    const float* Whead = (const float*)d_in[9];
    float* out = (float*)d_out;   // logits [B,T,V] then state [L,B,D]

    char* ws = (char*)d_ws;
    size_t off = 0;
    auto carve = [&](size_t bytes) {
        size_t o = off;
        off = (off + bytes + 255) & ~(size_t)255;
        return o;
    };
    float*     x     = (float*)    (ws + carve((size_t)BB * TT * DD * 4));
    _Float16*  Wpk   = (_Float16*) (ws + carve((size_t)LL * 32 * KC * 512 * 2));
    float*     hs    = (float*)    (ws + carve((size_t)BB * TT * DD * 4));
    _Float16*  hsA   = (_Float16*) (ws + carve((size_t)BB * TT * DD * 2));
    _Float16*  WBh   = (_Float16*) (ws + carve((size_t)VV * DD * 2));
    float*     sbufA = (float*)    (ws + carve((size_t)LL * BB * DD * 4));
    float*     sbufB = (float*)    (ws + carve((size_t)LL * BB * DD * 4));
    float*     h_buf = (float*)    (ws + carve((size_t)BB * DD * 4));
    unsigned*  bar   = (unsigned*) (ws + carve(256));

    float* out_state = out + (size_t)BB * TT * VV;

    (void)hipMemsetAsync(bar,   0, 4,                        stream);
    (void)hipMemsetAsync(sbufA, 0, (size_t)LL * BB * DD * 4, stream);
    (void)hipMemsetAsync(sbufB, 0, (size_t)LL * BB * DD * 4, stream);

    k_embed<<<(BB * TT * DD) / 256, 256, 0, stream>>>(ids, emb, x);
    k_pack_w<<<(LL * 32 * KC * 512) / 256, 256, 0, stream>>>(Win, Wrec, Wpk);
    k_pack_whead<<<((VV / 16) * KCH * 512) / 256, 256, 0, stream>>>(Whead, WBh);

    k_scan<<<NWG, 256, 0, stream>>>(x, Wpk, bias, tau, Avec, gamma, beta,
                                    sbufA, sbufB, h_buf, hs, out_state, bar);

    k_pack_hs<<<(BB * TT * DD) / 256, 256, 0, stream>>>(hs, hsA);
    k_gemm_head<<<dim3(VV / 256, (BB * TT) / 64), 256, 0, stream>>>(hsA, WBh,
                                                                    out);
    (void)in_sizes; (void)n_in; (void)out_size; (void)ws_size;
}
